// SelfAttention_59794534695248
// MI455X (gfx1250) — matrix operand order repaired and checked by reading, not compiled, not run on hardware
//
#include <hip/hip_runtime.h>

#define TB 256
#define BM 128
#define BN 128
#define BK 32
#define LDT 40  // padded LDS row stride (elements); 80B rows -> 16B aligned chunks

#define AS1 __attribute__((address_space(1)))
#define AS3 __attribute__((address_space(3)))

#if defined(__AMDGCN__) && __has_builtin(__builtin_amdgcn_global_load_async_to_lds_b128)
#define HAVE_ASYNC_LDS 1
#else
#define HAVE_ASYNC_LDS 0
#endif

typedef __attribute__((ext_vector_type(16))) __bf16 bf16x16;
typedef __attribute__((ext_vector_type(8)))  float  f32x8;
typedef int vint4 __attribute__((vector_size(4 * sizeof(int))));

struct alignas(16) U4 { unsigned int x, y, z, w; };
struct alignas(16) F4 { float x, y, z, w; };

union FragU { bf16x16 v; U4 u[2]; };

__device__ __forceinline__ unsigned short f2bf(float f) {
  union { float f; unsigned int u; } c; c.f = f;
  unsigned int r = c.u + 0x7FFFu + ((c.u >> 16) & 1u);  // round-to-nearest-even
  return (unsigned short)(r >> 16);
}

// 32B (16 bf16) global -> LDS copy; async-to-LDS on gfx1250 if available.
__device__ __forceinline__ void cp_g2l_32B(const unsigned short* g, unsigned short* l) {
#if HAVE_ASYNC_LDS
  AS1 vint4* gp = (AS1 vint4*)const_cast<unsigned short*>(g);
  AS3 vint4* lp = (AS3 vint4*)l;
  __builtin_amdgcn_global_load_async_to_lds_b128(gp, lp, 0, 0);
  __builtin_amdgcn_global_load_async_to_lds_b128(gp + 1, lp + 1, 0, 0);
#else
  *reinterpret_cast<U4*>(l)     = reinterpret_cast<const U4*>(g)[0];
  *reinterpret_cast<U4*>(l + 8) = reinterpret_cast<const U4*>(g)[1];
#endif
}

__device__ __forceinline__ void wait_async_lds() {
#if HAVE_ASYNC_LDS
#if __has_builtin(__builtin_amdgcn_s_wait_asynccnt)
  __builtin_amdgcn_s_wait_asynccnt(0);
#else
  asm volatile("s_wait_asynccnt 0x0" ::: "memory");
#endif
#endif
}

// One 128x128 block tile step over BK=32: 8 waves, each wave 32x64 (2x4 WMMA tiles).
__device__ __forceinline__ void wmma_tiles(const unsigned short* __restrict__ As,
                                           const unsigned short* __restrict__ Bs,
                                           f32x8 acc[2][4], int lane, int wm, int wn) {
  const int rr = lane & 15;   // A row / B col within 16-tile
  const int kg = lane >> 4;   // K-group select (CDNA5 16-bit A/B layouts)
  FragU a[2];
#pragma unroll
  for (int i = 0; i < 2; ++i) {
    const unsigned short* pa = As + (wm * 32 + i * 16 + rr) * LDT;
    a[i].u[0] = *reinterpret_cast<const U4*>(pa + kg * 8);        // K = 8*kg + 0..7
    a[i].u[1] = *reinterpret_cast<const U4*>(pa + 16 + kg * 8);   // K = 16 + 8*kg + 0..7
  }
  FragU b[4];
#pragma unroll
  for (int j = 0; j < 4; ++j) {
    const unsigned short* pb = Bs + (wn * 64 + j * 16 + rr) * LDT;  // K = 8*kg + 0..7, 16 + 8*kg + 0..7
    b[j].u[0] = *reinterpret_cast<const U4*>(pb + kg * 8);
    b[j].u[1] = *reinterpret_cast<const U4*>(pb + 16 + kg * 8);
  }
#pragma unroll
  for (int i = 0; i < 2; ++i)
#pragma unroll
    for (int j = 0; j < 4; ++j)
      acc[i][j] = __builtin_amdgcn_wmma_f32_16x16x32_bf16(
          false, a[i].v, false, b[j].v, (short)0, acc[i][j], false, false);
}

// ---------------- Kernel 1: Q/K/V = x*W + b (z selects matrix; V stored transposed) ----
__global__ __launch_bounds__(TB)
void qkv_proj_kernel(const float* __restrict__ x,
                     const float* __restrict__ Wq, const float* __restrict__ bq,
                     const float* __restrict__ Wk, const float* __restrict__ bk,
                     const float* __restrict__ Wv, const float* __restrict__ bv,
                     unsigned short* __restrict__ Q,
                     unsigned short* __restrict__ Kc,
                     unsigned short* __restrict__ Vt) {
  const int H = 1024, DK = 1024, S = 2048;
  __shared__ alignas(16) unsigned short As[2][BM * LDT];
  __shared__ alignas(16) unsigned short Bs[2][BN * LDT];

  const int which = blockIdx.z;
  const float* W    = (which == 0) ? Wq : (which == 1) ? Wk : Wv;
  const float* bias = (which == 0) ? bq : (which == 1) ? bk : bv;

  const int N0 = blockIdx.x * BN;
  const int M0 = blockIdx.y * BM;
  const int tid = threadIdx.x, lane = tid & 31, wid = tid >> 5;
  const int wm = wid >> 1, wn = wid & 1;

  const f32x8 vzero = {0.f, 0.f, 0.f, 0.f, 0.f, 0.f, 0.f, 0.f};
  f32x8 acc[2][4];
#pragma unroll
  for (int i = 0; i < 2; ++i)
#pragma unroll
    for (int j = 0; j < 4; ++j) acc[i][j] = vzero;

  const int rA = tid >> 1, kcA = (tid & 1) * 16;
  const int klB = tid >> 3, ncB = (tid & 7) * 16;
  const float* srcA0 = x + (size_t)(M0 + rA) * DK + kcA;
  const float* srcB0 = W + (size_t)klB * H + N0 + ncB;

  auto loadA = [&](int kb, float* g) {
    const float* s = srcA0 + kb * BK;
#pragma unroll
    for (int q4 = 0; q4 < 4; ++q4) {
      F4 t4 = reinterpret_cast<const F4*>(s)[q4];
      g[q4 * 4 + 0] = t4.x; g[q4 * 4 + 1] = t4.y; g[q4 * 4 + 2] = t4.z; g[q4 * 4 + 3] = t4.w;
    }
  };
  auto loadB = [&](int kb, float* g) {
    const float* s = srcB0 + (size_t)kb * BK * H;
#pragma unroll
    for (int q4 = 0; q4 < 4; ++q4) {
      F4 t4 = reinterpret_cast<const F4*>(s)[q4];
      g[q4 * 4 + 0] = t4.x; g[q4 * 4 + 1] = t4.y; g[q4 * 4 + 2] = t4.z; g[q4 * 4 + 3] = t4.w;
    }
  };
  auto writeA = [&](int buf, const float* g) {
    unsigned short* dst = &As[buf][rA * LDT + kcA];
#pragma unroll
    for (int i = 0; i < 16; ++i) dst[i] = f2bf(g[i]);
  };
  auto writeB = [&](int buf, const float* g) {  // W[k][n] -> LDS [n][k] transpose
#pragma unroll
    for (int i = 0; i < 16; ++i) Bs[buf][(ncB + i) * LDT + klB] = f2bf(g[i]);
  };

  {
    float ga[16], gb[16];
    loadA(0, ga); loadB(0, gb);
    writeA(0, ga); writeB(0, gb);
  }
  __syncthreads();

  const int NKB = DK / BK;
  for (int kb = 0; kb < NKB; ++kb) {
    const int cur = kb & 1;
    float ga[16], gb[16];
    const bool pf = (kb + 1 < NKB);
    if (pf) { loadA(kb + 1, ga); loadB(kb + 1, gb); }   // issue loads before math
    wmma_tiles(&As[cur][0], &Bs[cur][0], acc, lane, wm, wn);
    if (pf) { writeA(cur ^ 1, ga); writeB(cur ^ 1, gb); }
    __syncthreads();
  }

  const int nn = lane & 15, kg = lane >> 4;
#pragma unroll
  for (int i = 0; i < 2; ++i) {
#pragma unroll
    for (int j = 0; j < 4; ++j) {
      const int ncol = N0 + wn * 64 + j * 16 + nn;
      const float bb = bias[ncol];
#pragma unroll
      for (int v = 0; v < 8; ++v) {
        const int mrow = M0 + wm * 32 + i * 16 + kg * 8 + v;
        const unsigned short hv = f2bf(acc[i][j][v] + bb);
        if (which == 0) {
          Q[(size_t)mrow * H + ncol] = hv;
        } else if (which == 1) {
          Kc[(size_t)mrow * H + ncol] = hv;
        } else {  // V transposed: Vt[b][h][s]
          const int bidx = mrow >> 11, s = mrow & (S - 1);
          Vt[((size_t)bidx * H + ncol) * S + s] = hv;
        }
      }
    }
  }
}

// ---------------- Kernel 2: scores = Q * K^T / sqrt(H), f32 to workspace -------------
__global__ __launch_bounds__(TB)
void scores_kernel(const unsigned short* __restrict__ Q,
                   const unsigned short* __restrict__ Kc,
                   float* __restrict__ Sc) {
  const int H = 1024, S = 2048;
  __shared__ alignas(16) unsigned short As[2][BM * LDT];
  __shared__ alignas(16) unsigned short Bs[2][BN * LDT];
  const int b = blockIdx.z;
  const int N0 = blockIdx.x * BN, M0 = blockIdx.y * BM;
  const int tid = threadIdx.x, lane = tid & 31, wid = tid >> 5;
  const int wm = wid >> 1, wn = wid & 1;

  const f32x8 vzero = {0.f, 0.f, 0.f, 0.f, 0.f, 0.f, 0.f, 0.f};
  f32x8 acc[2][4];
#pragma unroll
  for (int i = 0; i < 2; ++i)
#pragma unroll
    for (int j = 0; j < 4; ++j) acc[i][j] = vzero;

  const int r = tid >> 1, kc = (tid & 1) * 16;
  const unsigned short* gA = Q + (size_t)b * S * H + (size_t)(M0 + r) * H + kc;
  const unsigned short* gB = Kc + (size_t)b * S * H + (size_t)(N0 + r) * H + kc;

  auto stage = [&](int kb, int buf) {
    cp_g2l_32B(gA + kb * BK, &As[buf][r * LDT + kc]);
    cp_g2l_32B(gB + kb * BK, &Bs[buf][r * LDT + kc]);
  };

  stage(0, 0);
  wait_async_lds();
  __syncthreads();

  const int NKB = H / BK;
  for (int kb = 0; kb < NKB; ++kb) {
    const int cur = kb & 1;
    if (kb + 1 < NKB) stage(kb + 1, cur ^ 1);  // async fill of other buffer
    wmma_tiles(&As[cur][0], &Bs[cur][0], acc, lane, wm, wn);
    wait_async_lds();
    __syncthreads();
  }

  float* ScB = Sc + (size_t)b * S * S;
  const int nn = lane & 15, kg = lane >> 4;
#pragma unroll
  for (int i = 0; i < 2; ++i)
#pragma unroll
    for (int j = 0; j < 4; ++j)
#pragma unroll
      for (int v = 0; v < 8; ++v)
        ScB[(size_t)(M0 + wm * 32 + i * 16 + kg * 8 + v) * S +
            (N0 + wn * 64 + j * 16 + nn)] = acc[i][j][v] * 0.03125f;  // 1/sqrt(1024)
}

// ---------------- Kernel 3: row-wise softmax in place on f32 scores ------------------
__global__ __launch_bounds__(TB)
void softmax_kernel(float* __restrict__ sc, int S) {
  __shared__ float red[TB];
  float* p = sc + (size_t)blockIdx.x * S;
  const int t = threadIdx.x;
  float m = -3.4e38f;
  for (int i = t; i < S; i += TB) m = fmaxf(m, p[i]);
  red[t] = m; __syncthreads();
  for (int o = TB / 2; o > 0; o >>= 1) {
    if (t < o) red[t] = fmaxf(red[t], red[t + o]);
    __syncthreads();
  }
  m = red[0]; __syncthreads();
  float s = 0.f;
  for (int i = t; i < S; i += TB) { float e = __expf(p[i] - m); p[i] = e; s += e; }
  red[t] = s; __syncthreads();
  for (int o = TB / 2; o > 0; o >>= 1) {
    if (t < o) red[t] += red[t + o];
    __syncthreads();
  }
  const float inv = 1.0f / red[0];
  for (int i = t; i < S; i += TB) p[i] *= inv;
}

// ---------------- Kernel 4: out = P * V  (P f32->bf16 on stage; B async from Vt) -----
__global__ __launch_bounds__(TB)
void attnout_kernel(const float* __restrict__ P,
                    const unsigned short* __restrict__ Vt,
                    float* __restrict__ O) {
  const int H = 1024, S = 2048;
  __shared__ alignas(16) unsigned short As[2][BM * LDT];
  __shared__ alignas(16) unsigned short Bs[2][BN * LDT];
  const int b = blockIdx.z;
  const int N0 = blockIdx.x * BN, M0 = blockIdx.y * BM;
  const int tid = threadIdx.x, lane = tid & 31, wid = tid >> 5;
  const int wm = wid >> 1, wn = wid & 1;

  const f32x8 vzero = {0.f, 0.f, 0.f, 0.f, 0.f, 0.f, 0.f, 0.f};
  f32x8 acc[2][4];
#pragma unroll
  for (int i = 0; i < 2; ++i)
#pragma unroll
    for (int j = 0; j < 4; ++j) acc[i][j] = vzero;

  const int r = tid >> 1, kc = (tid & 1) * 16;
  const float* gA = P + (size_t)b * S * S + (size_t)(M0 + r) * S + kc;
  const unsigned short* gB = Vt + (size_t)b * H * S + (size_t)(N0 + r) * S + kc;

  auto loadA = [&](int kb, float* g) {
    const float* s = gA + kb * BK;
#pragma unroll
    for (int q4 = 0; q4 < 4; ++q4) {
      F4 t4 = reinterpret_cast<const F4*>(s)[q4];
      g[q4 * 4 + 0] = t4.x; g[q4 * 4 + 1] = t4.y; g[q4 * 4 + 2] = t4.z; g[q4 * 4 + 3] = t4.w;
    }
  };
  auto writeA = [&](int buf, const float* g) {
    unsigned short* dst = &As[buf][r * LDT + kc];
#pragma unroll
    for (int i = 0; i < 16; ++i) dst[i] = f2bf(g[i]);
  };
  auto stageB = [&](int kb, int buf) {
    cp_g2l_32B(gB + kb * BK, &Bs[buf][r * LDT + kc]);
  };

  {
    float ga[16];
    loadA(0, ga); writeA(0, ga);
    stageB(0, 0);
  }
  wait_async_lds();
  __syncthreads();

  const int NKB = S / BK;
  for (int kb = 0; kb < NKB; ++kb) {
    const int cur = kb & 1;
    float ga[16];
    const bool pf = (kb + 1 < NKB);
    if (pf) { loadA(kb + 1, ga); stageB(kb + 1, cur ^ 1); }
    wmma_tiles(&As[cur][0], &Bs[cur][0], acc, lane, wm, wn);
    if (pf) writeA(cur ^ 1, ga);
    wait_async_lds();
    __syncthreads();
  }

  const int nn = lane & 15, kg = lane >> 4;
#pragma unroll
  for (int i = 0; i < 2; ++i)
#pragma unroll
    for (int j = 0; j < 4; ++j)
#pragma unroll
      for (int v = 0; v < 8; ++v)
        O[(size_t)b * S * H +
          (size_t)(M0 + wm * 32 + i * 16 + kg * 8 + v) * H +
          (N0 + wn * 64 + j * 16 + nn)] = acc[i][j][v];
}

extern "C" void kernel_launch(void* const* d_in, const int* in_sizes, int n_in,
                              void* d_out, int out_size, void* d_ws, size_t ws_size,
                              hipStream_t stream) {
  (void)in_sizes; (void)n_in; (void)out_size; (void)ws_size;
  const float* x  = (const float*)d_in[0];
  const float* Wq = (const float*)d_in[1];
  const float* bq = (const float*)d_in[2];
  const float* Wk = (const float*)d_in[3];
  const float* bk = (const float*)d_in[4];
  const float* Wv = (const float*)d_in[5];
  const float* bv = (const float*)d_in[6];
  float* out = (float*)d_out;

  char* ws = (char*)d_ws;
  unsigned short* Q  = (unsigned short*)(ws);                               // 16 MB bf16 [B*S][H]
  unsigned short* Kc = (unsigned short*)(ws + (size_t)16 * 1024 * 1024);    // 16 MB bf16 [B*S][H]
  unsigned short* Vt = (unsigned short*)(ws + (size_t)32 * 1024 * 1024);    // 16 MB bf16 [B][H][S]
  float*          Sc = (float*)(ws + (size_t)48 * 1024 * 1024);             // 64 MB f32  [B][S][S]

  dim3 blk(TB);
  // M=8192 rows (B*S), N=1024, z selects Wq/Wk/Wv
  qkv_proj_kernel<<<dim3(1024 / BN, 8192 / BM, 3), blk, 0, stream>>>(
      x, Wq, bq, Wk, bk, Wv, bv, Q, Kc, Vt);
  // per batch: M=N=2048, depth 1024
  scores_kernel<<<dim3(2048 / BN, 2048 / BM, 4), blk, 0, stream>>>(Q, Kc, Sc);
  // 8192 rows of length 2048
  softmax_kernel<<<dim3(8192), blk, 0, stream>>>(Sc, 2048);
  // per batch: M=2048, N=1024, depth 2048
  attnout_kernel<<<dim3(1024 / BN, 2048 / BM, 4), blk, 0, stream>>>(Sc, Vt, out);
}
